// LLMDTA_MoE_30210799960459
// MI455X (gfx1250) — matrix operand order repaired and checked by reading, not compiled, not run on hardware
//
#include <hip/hip_runtime.h>
#include <hip/hip_bf16.h>
#include <math.h>

typedef _Float16 v16h __attribute__((ext_vector_type(16)));
typedef _Float16 v8h  __attribute__((ext_vector_type(8)));
typedef float    v8f  __attribute__((ext_vector_type(8)));
typedef float    v4f  __attribute__((ext_vector_type(4)));

#define EPSV 1e-5f

__device__ __forceinline__ float act_apply(float v, int act) {
  if (act == 1) return v > 0.f ? v : 0.f;
  if (act == 2) return v > 0.f ? v : (expf(v) - 1.f);
  return v;
}

__device__ __forceinline__ v16h combine16(v8h lo, v8h hi) {
  return __builtin_shufflevector(lo, hi, 0, 1, 2, 3, 4, 5, 6, 7,
                                         8, 9, 10, 11, 12, 13, 14, 15);
}

// ---------------------------------------------------------------------------
// Generic WMMA GEMM, fully vectorized operand feeds.
//   A  : f16 [M, lda]  (K <= lda, K % 32 == 0, rows 16B-aligned)
//   Bt : f16 [N, K]    (transposed weights, contiguous K)
//   C  : act(A@B + bias) (+ optional BN epilogue), OT = float | f16
//   CT : optional extra transposed f16 store: CT[(row/Lq)*N + col][row%Lq]
// One 16x16 tile per wave, 4 waves per block.
// ---------------------------------------------------------------------------
template<typename OT>
__global__ void gemm_wmma(const _Float16* __restrict__ A, const _Float16* __restrict__ Bt,
                          const float* __restrict__ bias, OT* __restrict__ C,
                          int M, int N, int K, int lda, int ldc, int act,
                          const float* __restrict__ postg, const float* __restrict__ postb,
                          _Float16* __restrict__ CT, int Lq)
{
  const int wave = threadIdx.x >> 5;
  const int lane = threadIdx.x & 31;
  const int hi = lane >> 4, p = lane & 15;
  const int ntn = N >> 4;
  const int tile = blockIdx.x * 4 + wave;
  if (tile >= (M >> 4) * ntn) return;
  const int tm = tile / ntn, tn = tile - tm * ntn;
  const int row = tm * 16 + p;
  const int col = tn * 16 + p;
  const _Float16* arow = A + (size_t)row * lda;
  const _Float16* bcol = Bt + (size_t)col * K;
  v8f acc; for (int i = 0; i < 8; i++) acc[i] = 0.f;
  for (int kb = 0; kb < K; kb += 32) {
    __builtin_prefetch(arow + kb + 256, 0, 1);
    __builtin_prefetch(bcol + kb + 256, 0, 1);
    v8h a0 = *(const v8h*)(arow + kb + hi * 8);
    v8h a1 = *(const v8h*)(arow + kb + 16 + hi * 8);
    v8h b0 = *(const v8h*)(bcol + kb + hi * 8);
    v8h b1 = *(const v8h*)(bcol + kb + 16 + hi * 8);
    acc = __builtin_amdgcn_wmma_f32_16x16x32_f16(false, combine16(a0, a1),
                                                 false, combine16(b0, b1),
                                                 (short)0, acc, false, false);
  }
  const float inv = rsqrtf(1.f + EPSV);
#pragma unroll
  for (int r = 0; r < 8; r++) {
    int orow = tm * 16 + r + (hi << 3);
    float v = acc[r] + (bias ? bias[col] : 0.f);
    v = act_apply(v, act);
    if (postg) v = v * postg[col] * inv + postb[col];   // bn_eval
    C[(size_t)orow * ldc + col] = (OT)v;
    if (CT) {
      int bq = orow / Lq, q = orow - bq * Lq;
      CT[((size_t)bq * N + col) * Lq + q] = (_Float16)v;
    }
  }
}

// ---------------------------------------------------------------------------
// Implicit-GEMM conv1d(k=7,pad=3,Cin=128,Cout=256) + GLU + residual fused.
// Block = one 16-position tile x 4 cout-tiles. The 22x128 halo input tile is
// cooperatively staged to LDS in f16 once per block (padding pre-zeroed), so
// A-fragments are plain 16B ds_load_b128 reads shared by all 4 waves.
// Wt: f16 [cout][896] repacked weights (contiguous K).
// ---------------------------------------------------------------------------
__global__ void conv_glu_wmma(const float* __restrict__ Hin, const _Float16* __restrict__ Wt,
                              const float* __restrict__ bias, float* __restrict__ Hout,
                              int Btot, int L)
{
  __shared__ _Float16 sA[22 * 128];          // rows l0-3 .. l0+18, f16
  const int wave = threadIdx.x >> 5, lane = threadIdx.x & 31;
  const int hi = lane >> 4, p = lane & 15;
  const int tm = blockIdx.x >> 1;            // position tile
  const int tc = ((blockIdx.x & 1) << 2) + wave;  // cout tile 0..7
  const int m0 = tm << 4;
  const int b = m0 / L, l0 = m0 % L;         // 16-row tile stays in one batch
  const float* hb = Hin + (size_t)b * L * 128;

  // cooperative halo staging (coalesced f32 loads, zero-padded)
  for (int idx = threadIdx.x; idx < 22 * 128; idx += 128) {
    int r = idx >> 7, c = idx & 127;
    int ll = l0 + r - 3;
    sA[idx] = (ll >= 0 && ll < L) ? (_Float16)hb[(size_t)ll * 128 + c] : (_Float16)0.f;
  }
  __syncthreads();

  const int coutA = (tc << 4) + p, coutG = coutA + 128;
  const _Float16* bca = Wt + (size_t)coutA * 896;
  const _Float16* bcg = Wt + (size_t)coutG * 896;
  v8f accA, accG;
  for (int i = 0; i < 8; i++) { accA[i] = 0.f; accG[i] = 0.f; }
  for (int kb = 0; kb < 896; kb += 32) {
    __builtin_prefetch(bca + kb + 256, 0, 1);
    __builtin_prefetch(bcg + kb + 256, 0, 1);
    // A from LDS: k -> (tap, cin); LDS row = p + tap (halo handles padding)
    int k0 = kb + hi * 8;
    int k1 = kb + 16 + hi * 8;
    v8h a0 = *(const v8h*)(sA + (p + (k0 >> 7)) * 128 + (k0 & 127));
    v8h a1 = *(const v8h*)(sA + (p + (k1 >> 7)) * 128 + (k1 & 127));
    v16h af = combine16(a0, a1);
    v8h ba0 = *(const v8h*)(bca + kb + hi * 8);
    v8h ba1 = *(const v8h*)(bca + kb + 16 + hi * 8);
    v8h bg0 = *(const v8h*)(bcg + kb + hi * 8);
    v8h bg1 = *(const v8h*)(bcg + kb + 16 + hi * 8);
    accA = __builtin_amdgcn_wmma_f32_16x16x32_f16(false, af, false, combine16(ba0, ba1),
                                                  (short)0, accA, false, false);
    accG = __builtin_amdgcn_wmma_f32_16x16x32_f16(false, af, false, combine16(bg0, bg1),
                                                  (short)0, accG, false, false);
  }
  const float scale = 0.70710678118654752f;
#pragma unroll
  for (int r = 0; r < 8; r++) {
    int l = l0 + r + (hi << 3);
    float a = accA[r] + bias[coutA];
    float g = accG[r] + bias[coutG];
    float sg = 1.f / (1.f + expf(-g));
    size_t idx = ((size_t)b * L + l) * 128 + coutA;
    Hout[idx] = (a * sg + Hin[idx]) * scale;   // residual in f32
  }
}

// ---------------------------------------------------------------------------
// att[b,h,v,q] = sum_k (v_[b,v,k]*h_mat[h,k]) * q_[b,q,k] + h_bias[h]
// All operands f16, fully contiguous b128 loads.
// ---------------------------------------------------------------------------
__global__ void att_wmma(const _Float16* __restrict__ Vv, const _Float16* __restrict__ Qv,
                         const _Float16* __restrict__ hm16, const float* __restrict__ hbias,
                         _Float16* __restrict__ att, int V, int Q)
{
  const int wave = threadIdx.x >> 5, lane = threadIdx.x & 31;
  const int hi = lane >> 4, p = lane & 15;
  const int bh = blockIdx.y;                 // b*2 + h
  const int b = bh >> 1, h = bh & 1;
  const int qt = Q >> 4;
  const int tile = blockIdx.x * 4 + wave;
  if (tile >= (V >> 4) * qt) return;
  const int tm = tile / qt, tn = tile - tm * qt;
  const _Float16* arow = Vv + ((size_t)b * V + tm * 16 + p) * 768;
  const _Float16* hrow = hm16 + h * 768;
  const _Float16* bcol = Qv + ((size_t)b * Q + tn * 16 + p) * 768;
  v8f acc; for (int i = 0; i < 8; i++) acc[i] = 0.f;
  for (int kb = 0; kb < 768; kb += 32) {
    __builtin_prefetch(arow + kb + 256, 0, 1);
    __builtin_prefetch(bcol + kb + 256, 0, 1);
    v8h a0 = *(const v8h*)(arow + kb + hi * 8)      * *(const v8h*)(hrow + kb + hi * 8);
    v8h a1 = *(const v8h*)(arow + kb + 16 + hi * 8) * *(const v8h*)(hrow + kb + 16 + hi * 8);
    v8h b0 = *(const v8h*)(bcol + kb + hi * 8);
    v8h b1 = *(const v8h*)(bcol + kb + 16 + hi * 8);
    acc = __builtin_amdgcn_wmma_f32_16x16x32_f16(false, combine16(a0, a1),
                                                 false, combine16(b0, b1),
                                                 (short)0, acc, false, false);
  }
  const float hb = hbias[h];
#pragma unroll
  for (int r = 0; r < 8; r++) {
    int v = tm * 16 + r + (hi << 3);
    att[(((size_t)bh) * V + v) * Q + tn * 16 + p] = (_Float16)(acc[r] + hb);
  }
}

// ---------------------------------------------------------------------------
// fuse[b,k] = sum_h sum_v v_[b,v,k] * (att[b,h] @ q_[b])[v,k]
// A = att rows (contiguous), B = qT[b][k][q] (contiguous) -> b128 feeds.
// ---------------------------------------------------------------------------
__global__ void fuse_wmma(const _Float16* __restrict__ att, const _Float16* __restrict__ qT,
                          const _Float16* __restrict__ Vv, float* __restrict__ fuse,
                          int V, int Q)
{
  __shared__ float red[16];
  const int wave = threadIdx.x >> 5, lane = threadIdx.x & 31;
  const int hi = lane >> 4, p = lane & 15;
  const int kt = blockIdx.x;              // 0..47
  const int b = blockIdx.y;
  if (threadIdx.x < 16) red[threadIdx.x] = 0.f;
  __syncthreads();
  const int vt = wave;                    // 8 waves cover V=128
  const _Float16* bcol = qT + ((size_t)b * 768 + kt * 16 + p) * Q;
  float partial = 0.f;
  for (int h = 0; h < 2; h++) {
    v8f acc; for (int i = 0; i < 8; i++) acc[i] = 0.f;
    const _Float16* arow = att + (((size_t)(b * 2 + h)) * V + vt * 16 + p) * Q;
    for (int kb = 0; kb < Q; kb += 32) {
      v8h a0 = *(const v8h*)(arow + kb + hi * 8);
      v8h a1 = *(const v8h*)(arow + kb + 16 + hi * 8);
      v8h b0 = *(const v8h*)(bcol + kb + hi * 8);
      v8h b1 = *(const v8h*)(bcol + kb + 16 + hi * 8);
      acc = __builtin_amdgcn_wmma_f32_16x16x32_f16(false, combine16(a0, a1),
                                                   false, combine16(b0, b1),
                                                   (short)0, acc, false, false);
    }
#pragma unroll
    for (int r = 0; r < 8; r++) {
      int v = vt * 16 + r + (hi << 3);
      partial += (float)Vv[((size_t)b * V + v) * 768 + kt * 16 + p] * acc[r];
    }
  }
  atomicAdd(&red[p], partial);
  __syncthreads();
  if (threadIdx.x < 16) fuse[(size_t)b * 768 + kt * 16 + threadIdx.x] = red[threadIdx.x];
}

// --------------------------- repack kernels --------------------------------
// out[n*Kp + k] = (k<K) ? in[k*N + n] : 0   (weight transpose + f16 + K pad)
__global__ void repack_T(const float* __restrict__ in, _Float16* __restrict__ out,
                         int K, int N, int Kp)
{
  int idx = blockIdx.x * 256 + threadIdx.x;
  if (idx >= N * Kp) return;
  int n = idx / Kp, k = idx - n * Kp;
  out[idx] = (k < K) ? (_Float16)in[(size_t)k * N + n] : (_Float16)0.f;
}

// activations: out[r*Kp + k] = (k<K) ? in[r*K + k] : 0
__global__ void cast_pad_f16(const float* __restrict__ in, _Float16* __restrict__ out,
                             int rows, int K, int Kp)
{
  int idx = blockIdx.x * 256 + threadIdx.x;
  if (idx >= rows * Kp) return;
  int r = idx / Kp, k = idx - r * Kp;
  out[idx] = (k < K) ? (_Float16)in[(size_t)r * K + k] : (_Float16)0.f;
}

// conv weights [256][128][7] -> [256][896] f16, k = tap*128+cin
__global__ void repack_convw(const float* __restrict__ in, _Float16* __restrict__ out)
{
  int idx = blockIdx.x * 256 + threadIdx.x;
  if (idx >= 256 * 896) return;
  int cout = idx / 896, k = idx - cout * 896;
  int tap = k >> 7, cin = k & 127;
  out[idx] = (_Float16)in[((size_t)cout * 128 + cin) * 7 + tap];
}

__global__ void cast_f16(const float* __restrict__ in, _Float16* __restrict__ out, int n)
{
  int i = blockIdx.x * 256 + threadIdx.x;
  if (i < n) out[i] = (_Float16)in[i];
}

// --------------------------- VALU tail kernels -----------------------------
template<typename OT>
__global__ void row_ln(const float* __restrict__ X, OT* __restrict__ Y,
                       const float* __restrict__ g, const float* __restrict__ bta,
                       int rows, int W, int act)
{
  const int row = blockIdx.x * 4 + (threadIdx.x >> 5);
  const int lane = threadIdx.x & 31;
  if (row >= rows) return;
  const float* x = X + (size_t)row * W;
  float s = 0.f, s2 = 0.f;
  for (int i = lane; i < W; i += 32) { float v = x[i]; s += v; s2 += v * v; }
  for (int o = 16; o > 0; o >>= 1) { s += __shfl_xor(s, o, 32); s2 += __shfl_xor(s2, o, 32); }
  float mean = s / W;
  float var = s2 / W - mean * mean;
  float inv = rsqrtf(var + EPSV);
  OT* y = Y + (size_t)row * W;
  for (int i = lane; i < W; i += 32) {
    float v = (x[i] - mean) * inv * g[i] + bta[i];
    y[i] = (OT)act_apply(v, act);
  }
}

__global__ void maxpool_l(const float* __restrict__ H, float* __restrict__ P, int L)
{
  int b = blockIdx.x, c = threadIdx.x;   // 128 channels
  float m = -3.402823466e38f;
  for (int l = 0; l < L; l++) m = fmaxf(m, H[((size_t)b * L + l) * 128 + c]);
  P[b * 128 + c] = m;
}

__global__ void fuse_pool_bn(const float* __restrict__ F, const float* __restrict__ g,
                             const float* __restrict__ bt, _Float16* __restrict__ P)
{
  int b = blockIdx.x, c = threadIdx.x;   // 256
  float s = F[b * 768 + 3 * c] + F[b * 768 + 3 * c + 1] + F[b * 768 + 3 * c + 2];
  P[b * 256 + c] = (_Float16)(s * g[c] * rsqrtf(1.f + EPSV) + bt[c]);
}

__global__ void concat_pool(const float* __restrict__ A, const float* __restrict__ Bp,
                            _Float16* __restrict__ C)
{
  int b = blockIdx.x, c = threadIdx.x;   // 256
  C[b * 256 + c] = (_Float16)((c < 128) ? A[b * 128 + c] : Bp[b * 128 + c - 128]);
}

__global__ void vec_add_f16(const float* __restrict__ A, const float* __restrict__ B,
                            _Float16* __restrict__ C, int n)
{
  int i = blockIdx.x * 256 + threadIdx.x;
  if (i < n) C[i] = (_Float16)(A[i] + B[i]);
}

__global__ void gate_logits(const float* __restrict__ T, const float* __restrict__ w,
                            const float* __restrict__ bias, float* __restrict__ Out, int Bn)
{
  int id = blockIdx.x * 128 + threadIdx.x;
  if (id >= Bn * 4) return;
  int b = id >> 2, e = id & 3;
  float s = bias[e];
  for (int j = 0; j < 256; j++) s += T[b * 256 + j] * w[j * 4 + e];
  Out[id] = s;
}

__global__ void expert_out(const float* __restrict__ H1, const float* __restrict__ w2,
                           const float* __restrict__ b2, float* __restrict__ Out, int Bn)
{
  int id = blockIdx.x * 128 + threadIdx.x;   // e*Bn + b
  if (id >= 4 * Bn) return;
  int e = id / Bn, b = id % Bn;
  float s = b2[e];
  const float* h = H1 + ((size_t)e * Bn + b) * 512;
  const float* w = w2 + e * 512;
  for (int j = 0; j < 512; j++) s += h[j] * w[j];
  Out[id] = s;
}

__global__ void moe_final(const float* __restrict__ logits, const float* __restrict__ eout,
                          float* __restrict__ out, int Bn)
{
  int b = blockIdx.x * 128 + threadIdx.x;
  if (b >= Bn) return;
  float l[4];
  for (int e = 0; e < 4; e++) l[e] = logits[b * 4 + e];
  float mx = fmaxf(fmaxf(l[0], l[1]), fmaxf(l[2], l[3]));
  float r[4], ssum = 0.f;
  for (int e = 0; e < 4; e++) { r[e] = expf(l[e] - mx); ssum += r[e]; }
  for (int e = 0; e < 4; e++) { r[e] /= ssum; out[Bn + b * 4 + e] = r[e]; }
  int i0 = 0;
  for (int e = 1; e < 4; e++) if (r[e] > r[i0]) i0 = e;
  int i1 = -1;
  for (int e = 0; e < 4; e++) { if (e == i0) continue; if (i1 < 0 || r[e] > r[i1]) i1 = e; }
  float m2 = fmaxf(r[i0], r[i1]);
  float q0 = expf(r[i0] - m2), q1 = expf(r[i1] - m2);
  float n0 = q0 / (q0 + q1), n1 = q1 / (q0 + q1);
  out[b] = eout[i0 * Bn + b] * n0 + eout[i1 * Bn + b] * n1;
}

// ---------------------------------------------------------------------------
extern "C" void kernel_launch(void* const* d_in, const int* in_sizes, int n_in,
                              void* d_out, int out_size, void* d_ws, size_t ws_size,
                              hipStream_t stream)
{
  (void)in_sizes; (void)n_in; (void)out_size; (void)ws_size;
  const int B = 128, LD = 128, LP = 512;
  const float* drug_mat = (const float*)d_in[1];
  const float* prot_mat = (const float*)d_in[4];
  int i = 6;   // params pytree leaves, sorted-key order
  const float* ban_bn_b = (const float*)d_in[i++];
  const float* ban_bn_g = (const float*)d_in[i++];
  const float* h_bias   = (const float*)d_in[i++];
  const float* h_mat    = (const float*)d_in[i++];
  const float* qb       = (const float*)d_in[i++];
  const float* qw       = (const float*)d_in[i++];
  const float* vb       = (const float*)d_in[i++];
  const float* vw       = (const float*)d_in[i++];
  const float* bn_b     = (const float*)d_in[i++];
  const float* bn_g     = (const float*)d_in[i++];
  const float* d_conv_b = (const float*)d_in[i++];
  const float* d_conv_w = (const float*)d_in[i++];
  const float* d_fc_b   = (const float*)d_in[i++];
  const float* d_fc_w   = (const float*)d_in[i++];
  const float* d_ln_b   = (const float*)d_in[i++];
  const float* d_ln_g   = (const float*)d_in[i++];
  const float* e_b1     = (const float*)d_in[i++];
  const float* e_b2     = (const float*)d_in[i++];
  const float* e_ln_b   = (const float*)d_in[i++];
  const float* e_ln_g   = (const float*)d_in[i++];
  const float* e_w1     = (const float*)d_in[i++];
  const float* e_w2     = (const float*)d_in[i++];
  const float* g_b1     = (const float*)d_in[i++];
  const float* g_b2     = (const float*)d_in[i++];
  const float* g_b3     = (const float*)d_in[i++];
  const float* g_ln1_b  = (const float*)d_in[i++];
  const float* g_ln1_g  = (const float*)d_in[i++];
  const float* g_ln2_b  = (const float*)d_in[i++];
  const float* g_ln2_g  = (const float*)d_in[i++];
  const float* g_w1     = (const float*)d_in[i++];
  const float* g_w2     = (const float*)d_in[i++];
  const float* g_w3     = (const float*)d_in[i++];
  const float* post_b   = (const float*)d_in[i++];
  const float* post_w   = (const float*)d_in[i++];
  const float* pre_b    = (const float*)d_in[i++];
  const float* pre_w    = (const float*)d_in[i++];
  const float* p_conv_b = (const float*)d_in[i++];
  const float* p_conv_w = (const float*)d_in[i++];
  const float* p_fc_b   = (const float*)d_in[i++];
  const float* p_fc_w   = (const float*)d_in[i++];
  const float* p_ln_b   = (const float*)d_in[i++];
  const float* p_ln_g   = (const float*)d_in[i++];

  char* base = (char*)d_ws;
  size_t off = 0;
  auto alloc = [&](size_t bytes) -> void* {
    off = (off + 255) & ~(size_t)255;
    void* p = base + off;
    off += bytes;
    return p;
  };
  const int MD = B * LD;   // 16384
  const int MP = B * LP;   // 65536
  // f32 activations
  float* dh0 = (float*)alloc((size_t)MD * 128 * 4);
  float* dh1 = (float*)alloc((size_t)MD * 128 * 4);
  float* ph0 = (float*)alloc((size_t)MP * 128 * 4);
  float* ph1 = (float*)alloc((size_t)MP * 128 * 4);
  float* dpool = (float*)alloc((size_t)B * 128 * 4);
  float* ppool = (float*)alloc((size_t)B * 128 * 4);
  float* fuseb = (float*)alloc((size_t)B * 768 * 4);
  float* hpre  = (float*)alloc((size_t)B * 1024 * 4);
  float* hpost = (float*)alloc((size_t)B * 1024 * 4);
  float* t1raw = (float*)alloc((size_t)B * 512 * 4);
  float* t2raw = (float*)alloc((size_t)B * 256 * 4);
  float* t2    = (float*)alloc((size_t)B * 256 * 4);
  float* t3    = (float*)alloc((size_t)B * 4 * 4);
  float* eh1   = (float*)alloc((size_t)4 * B * 512 * 4);
  float* eoutb = (float*)alloc((size_t)4 * B * 4);
  // f16 activations
  _Float16* a_d16 = (_Float16*)alloc((size_t)MD * 320 * 2);
  _Float16* a_p16 = (_Float16*)alloc((size_t)MP * 128 * 2);
  _Float16* demb  = (_Float16*)alloc((size_t)MD * 128 * 2);
  _Float16* pemb  = (_Float16*)alloc((size_t)MP * 128 * 2);
  _Float16* vbuf  = (_Float16*)alloc((size_t)MD * 768 * 2);
  _Float16* qbuf  = (_Float16*)alloc((size_t)MP * 768 * 2);
  _Float16* qbufT = (_Float16*)alloc((size_t)B * 768 * LP * 2);
  _Float16* attb  = (_Float16*)alloc((size_t)B * 2 * LD * LP * 2);
  _Float16* cat16 = (_Float16*)alloc((size_t)B * 256 * 2);
  _Float16* bp16  = (_Float16*)alloc((size_t)B * 256 * 2);
  _Float16* x16   = (_Float16*)alloc((size_t)B * 1024 * 2);
  _Float16* t116  = (_Float16*)alloc((size_t)B * 512 * 2);
  // f16 transposed weights
  _Float16* wT_fc_d = (_Float16*)alloc((size_t)128 * 320 * 2);
  _Float16* wT_fc_p = (_Float16*)alloc((size_t)128 * 128 * 2);
  _Float16* wT_cv_d = (_Float16*)alloc((size_t)3 * 256 * 896 * 2);
  _Float16* wT_cv_p = (_Float16*)alloc((size_t)3 * 256 * 896 * 2);
  _Float16* wT_v    = (_Float16*)alloc((size_t)768 * 128 * 2);
  _Float16* wT_q    = (_Float16*)alloc((size_t)768 * 128 * 2);
  _Float16* hm16    = (_Float16*)alloc((size_t)2 * 768 * 2);
  _Float16* wT_pre  = (_Float16*)alloc((size_t)1024 * 256 * 2);
  _Float16* wT_post = (_Float16*)alloc((size_t)1024 * 256 * 2);
  _Float16* wT_g1   = (_Float16*)alloc((size_t)512 * 1024 * 2);
  _Float16* wT_g2   = (_Float16*)alloc((size_t)256 * 512 * 2);
  _Float16* wT_e1   = (_Float16*)alloc((size_t)4 * 512 * 1024 * 2);

  auto nblk = [](int tiles) { return (tiles + 3) / 4; };
  auto eblk = [](long n) { return (int)((n + 255) / 256); };

  // ---------------- repacks (weights -> f16 [N][K], inputs -> padded f16) --
  repack_T<<<eblk(128 * 320), 256, 0, stream>>>(d_fc_w, wT_fc_d, 300, 128, 320);
  repack_T<<<eblk(128 * 128), 256, 0, stream>>>(p_fc_w, wT_fc_p, 100, 128, 128);
  cast_pad_f16<<<eblk((long)MD * 320), 256, 0, stream>>>(drug_mat, a_d16, MD, 300, 320);
  cast_pad_f16<<<eblk((long)MP * 128), 256, 0, stream>>>(prot_mat, a_p16, MP, 100, 128);
  for (int l = 0; l < 3; l++) {
    repack_convw<<<eblk(256 * 896), 256, 0, stream>>>(d_conv_w + (size_t)l * 229376,
                                                      wT_cv_d + (size_t)l * 229376);
    repack_convw<<<eblk(256 * 896), 256, 0, stream>>>(p_conv_w + (size_t)l * 229376,
                                                      wT_cv_p + (size_t)l * 229376);
  }
  repack_T<<<eblk(768 * 128), 256, 0, stream>>>(vw, wT_v, 128, 768, 128);
  repack_T<<<eblk(768 * 128), 256, 0, stream>>>(qw, wT_q, 128, 768, 128);
  cast_f16<<<eblk(1536), 256, 0, stream>>>(h_mat, hm16, 1536);
  repack_T<<<eblk(1024 * 256), 256, 0, stream>>>(pre_w, wT_pre, 256, 1024, 256);
  repack_T<<<eblk(1024 * 256), 256, 0, stream>>>(post_w, wT_post, 256, 1024, 256);
  repack_T<<<eblk(512 * 1024), 256, 0, stream>>>(g_w1, wT_g1, 1024, 512, 1024);
  repack_T<<<eblk(256 * 512), 256, 0, stream>>>(g_w2, wT_g2, 512, 256, 512);
  for (int e = 0; e < 4; e++)
    repack_T<<<eblk(512 * 1024), 256, 0, stream>>>(e_w1 + (size_t)e * 524288,
                                                   wT_e1 + (size_t)e * 524288, 1024, 512, 1024);

  // ---------------- drug encoder ----------------
  gemm_wmma<float><<<nblk((MD / 16) * 8), 128, 0, stream>>>(
      a_d16, wT_fc_d, d_fc_b, dh0, MD, 128, 320, 320, 128, 0, nullptr, nullptr, nullptr, 0);
  conv_glu_wmma<<<(MD / 16) * 2, 128, 0, stream>>>(dh0, wT_cv_d,          d_conv_b,       dh1, B, LD);
  conv_glu_wmma<<<(MD / 16) * 2, 128, 0, stream>>>(dh1, wT_cv_d + 229376, d_conv_b + 256, dh0, B, LD);
  conv_glu_wmma<<<(MD / 16) * 2, 128, 0, stream>>>(dh0, wT_cv_d + 458752, d_conv_b + 512, dh1, B, LD);
  maxpool_l<<<B, 128, 0, stream>>>(dh1, dpool, LD);
  row_ln<_Float16><<<(MD + 3) / 4, 128, 0, stream>>>(dh1, demb, d_ln_g, d_ln_b, MD, 128, 0);

  // ---------------- protein encoder ----------------
  gemm_wmma<float><<<nblk((MP / 16) * 8), 128, 0, stream>>>(
      a_p16, wT_fc_p, p_fc_b, ph0, MP, 128, 128, 128, 128, 0, nullptr, nullptr, nullptr, 0);
  conv_glu_wmma<<<(MP / 16) * 2, 128, 0, stream>>>(ph0, wT_cv_p,          p_conv_b,       ph1, B, LP);
  conv_glu_wmma<<<(MP / 16) * 2, 128, 0, stream>>>(ph1, wT_cv_p + 229376, p_conv_b + 256, ph0, B, LP);
  conv_glu_wmma<<<(MP / 16) * 2, 128, 0, stream>>>(ph0, wT_cv_p + 458752, p_conv_b + 512, ph1, B, LP);
  maxpool_l<<<B, 128, 0, stream>>>(ph1, ppool, LP);
  row_ln<_Float16><<<(MP + 3) / 4, 128, 0, stream>>>(ph1, pemb, p_ln_g, p_ln_b, MP, 128, 0);

  // ---------------- BAN ----------------
  gemm_wmma<_Float16><<<nblk((MD / 16) * 48), 128, 0, stream>>>(
      demb, wT_v, vb, vbuf, MD, 768, 128, 128, 768, 1, nullptr, nullptr, nullptr, 0);
  gemm_wmma<_Float16><<<nblk((MP / 16) * 48), 128, 0, stream>>>(
      pemb, wT_q, qb, qbuf, MP, 768, 128, 128, 768, 1, nullptr, nullptr, qbufT, LP);
  att_wmma<<<dim3(nblk((LD / 16) * (LP / 16)), B * 2), 128, 0, stream>>>(
      vbuf, qbuf, hm16, h_bias, attb, LD, LP);
  fuse_wmma<<<dim3(48, B), 256, 0, stream>>>(attb, qbufT, vbuf, fuseb, LD, LP);
  fuse_pool_bn<<<B, 256, 0, stream>>>(fuseb, ban_bn_g, ban_bn_b, bp16);

  // ---------------- head ----------------
  concat_pool<<<B, 256, 0, stream>>>(dpool, ppool, cat16);
  gemm_wmma<float><<<nblk((B / 16) * 64), 128, 0, stream>>>(
      cat16, wT_pre, pre_b, hpre, B, 1024, 256, 256, 1024, 2, bn_g, bn_b, nullptr, 0);
  gemm_wmma<float><<<nblk((B / 16) * 64), 128, 0, stream>>>(
      bp16, wT_post, post_b, hpost, B, 1024, 256, 256, 1024, 2, nullptr, nullptr, nullptr, 0);
  vec_add_f16<<<eblk(B * 1024), 256, 0, stream>>>(hpre, hpost, x16, B * 1024);

  gemm_wmma<float><<<nblk((B / 16) * 32), 128, 0, stream>>>(
      x16, wT_g1, g_b1, t1raw, B, 512, 1024, 1024, 512, 0, nullptr, nullptr, nullptr, 0);
  row_ln<_Float16><<<(B + 3) / 4, 128, 0, stream>>>(t1raw, t116, g_ln1_g, g_ln1_b, B, 512, 1);
  gemm_wmma<float><<<nblk((B / 16) * 16), 128, 0, stream>>>(
      t116, wT_g2, g_b2, t2raw, B, 256, 512, 512, 256, 0, nullptr, nullptr, nullptr, 0);
  row_ln<float><<<(B + 3) / 4, 128, 0, stream>>>(t2raw, t2, g_ln2_g, g_ln2_b, B, 256, 1);
  gate_logits<<<(B * 4 + 127) / 128, 128, 0, stream>>>(t2, g_w3, g_b3, t3, B);

  for (int e = 0; e < 4; e++) {
    gemm_wmma<float><<<nblk((B / 16) * 32), 128, 0, stream>>>(
        x16, wT_e1 + (size_t)e * 524288, e_b1 + e * 512, eh1 + (size_t)e * B * 512,
        B, 512, 1024, 1024, 512, 0, nullptr, nullptr, nullptr, 0);
    row_ln<float><<<(B + 3) / 4, 128, 0, stream>>>(
        eh1 + (size_t)e * B * 512, eh1 + (size_t)e * B * 512,
        e_ln_g + e * 512, e_ln_b + e * 512, B, 512, 2);
  }
  expert_out<<<(4 * B + 127) / 128, 128, 0, stream>>>(eh1, e_w2, e_b2, eoutb, B);
  moe_final<<<(B + 127) / 128, 128, 0, stream>>>(t3, eoutb, (float*)d_out, B);
}
